// SelfAttention_66443144069420
// MI455X (gfx1250) — matrix operand order repaired and checked by reading, not compiled, not run on hardware
//
#include <hip/hip_runtime.h>
#include <hip/hip_bf16.h>
#include <math.h>

// ---------------------------------------------------------------------------
// Problem constants (from reference): B=4, S=1024, DIM=2048, H=16, KV=4,
// HD=128, ROPE_DIM=64, causal mask, GQA repeat = 4.
// ---------------------------------------------------------------------------

typedef __bf16 bf16_t;
typedef __attribute__((ext_vector_type(8)))  __bf16 v8bf;
typedef __attribute__((ext_vector_type(16))) __bf16 v16bf;
typedef __attribute__((ext_vector_type(8)))  float  v8f;

__device__ __forceinline__ v16bf cat8(v8bf lo, v8bf hi) {
  return __builtin_shufflevector(lo, hi, 0,1,2,3,4,5,6,7,8,9,10,11,12,13,14,15);
}

__device__ __forceinline__ v8f vzero8() {
  v8f v;
#pragma unroll
  for (int i = 0; i < 8; ++i) v[i] = 0.0f;
  return v;
}

// Async copy 16B global -> LDS (gfx1250, tracked by ASYNCcnt).
// lds_off: byte offset in LDS (low 32 bits of the flat shared-aperture addr).
__device__ __forceinline__ void async_g2l_b128(unsigned lds_off,
                                               const void* gptr) {
  unsigned long long ga = (unsigned long long)(uintptr_t)gptr;
  asm volatile("global_load_async_to_lds_b128 %0, %1, off"
               :: "v"(lds_off), "v"(ga)
               : "memory");
}

__device__ __forceinline__ void wait_asynccnt0() {
  asm volatile("s_wait_asynccnt 0x0" ::: "memory");
}

// ---------------------------------------------------------------------------
// fp32 -> bf16 cast (4 elems / thread)
// ---------------------------------------------------------------------------
__global__ void cast_f32_bf16(const float* __restrict__ in,
                              bf16_t* __restrict__ out, long n) {
  long i = ((long)blockIdx.x * blockDim.x + threadIdx.x) * 4;
  if (i + 3 < n) {
    float4 v = *(const float4*)(in + i);
    out[i + 0] = (bf16_t)v.x;
    out[i + 1] = (bf16_t)v.y;
    out[i + 2] = (bf16_t)v.z;
    out[i + 3] = (bf16_t)v.w;
  } else {
    for (long j = i; j < n; ++j) out[j] = (bf16_t)in[j];
  }
}

// ---------------------------------------------------------------------------
// C[M,N] (fp32) = A[M,K] (bf16, row-major) * W[N,K]^T (bf16, row-major)
// Block = 128 threads = 4 waves; block tile 128x128; wave tile 64x64.
// Per wave per k-step (K chunk 32): 4 A-frags, 4 B-frags, 16 WMMAs.
// ---------------------------------------------------------------------------
__global__ void __launch_bounds__(128)
gemm_bf16_nt(const bf16_t* __restrict__ A, const bf16_t* __restrict__ W,
             float* __restrict__ C, int M, int N, int K) {
  const int lane = threadIdx.x & 31;
  const int wave = threadIdx.x >> 5;
  const int l15  = lane & 15;
  const int hi   = lane >> 4;
  const int n0 = blockIdx.x * 128 + (wave & 1) * 64;
  const int m0 = blockIdx.y * 128 + (wave >> 1) * 64;
  (void)M;

  v8f acc[4][4];
#pragma unroll
  for (int f = 0; f < 4; ++f)
#pragma unroll
    for (int g = 0; g < 4; ++g) acc[f][g] = vzero8();

  for (int kk = 0; kk < K; kk += 32) {
    v16bf af[4], bw[4];
#pragma unroll
    for (int f = 0; f < 4; ++f) {
      // A-frag 16x32: lane row = l15; K chunks [hi*8..+7] and [16+hi*8..+7]
      const bf16_t* ap = A + (size_t)(m0 + 16 * f + l15) * K + kk + hi * 8;
      af[f] = cat8(*(const v8bf*)ap, *(const v8bf*)(ap + 16));
    }
#pragma unroll
    for (int g = 0; g < 4; ++g) {
      // B-frag 32x16: lane col = l15; K chunks [hi*8..+7] and [16+hi*8..+7]
      const bf16_t* bp = W + (size_t)(n0 + 16 * g + l15) * K + kk + hi * 8;
      bw[g] = cat8(*(const v8bf*)bp, *(const v8bf*)(bp + 16));
    }
#pragma unroll
    for (int f = 0; f < 4; ++f)
#pragma unroll
      for (int g = 0; g < 4; ++g)
        acc[f][g] = __builtin_amdgcn_wmma_f32_16x16x32_bf16(
            false, af[f], false, bw[g], (short)0, acc[f][g], false, false);
  }

#pragma unroll
  for (int f = 0; f < 4; ++f)
#pragma unroll
    for (int g = 0; g < 4; ++g)
#pragma unroll
      for (int i = 0; i < 8; ++i) {
        int row = m0 + 16 * f + 8 * hi + i;
        int col = n0 + 16 * g + l15;
        C[(size_t)row * N + col] = acc[f][g][i];
      }
}

// ---------------------------------------------------------------------------
// RMS-norm + partial RoPE + layout shuffle.
// qkv: [B*S, 3072] fp32 (q 0..2047 | k 2048..2559 | v 2560..3071)
// slot 0..15  -> q head  (norm+rope) -> qh bf16 [B,16,S,128]
// slot 16..19 -> k head  (norm+rope) -> kh bf16 [B,4,S,128]
// slot 20..23 -> v head             -> vf fp32 [B,4,S,128], vT bf16 [B,4,128,S]
// ---------------------------------------------------------------------------
__global__ void __launch_bounds__(128)
normrope_kernel(const float* __restrict__ qkv, bf16_t* __restrict__ qh,
                bf16_t* __restrict__ kh, bf16_t* __restrict__ vT,
                float* __restrict__ vf) {
  const int r    = blockIdx.x;   // b*S + s
  const int slot = blockIdx.y;   // 0..23
  const int d    = threadIdx.x;  // 0..127
  const int b = r >> 10;
  const int s = r & 1023;

  __shared__ float red[128];
  __shared__ float sv[128];

  float val = qkv[(size_t)r * 3072 + (size_t)slot * 128 + d];

  if (slot < 20) {
    red[d] = val * val;
    __syncthreads();
    for (int st = 64; st > 0; st >>= 1) {
      if (d < st) red[d] += red[d + st];
      __syncthreads();
    }
    float rs = rsqrtf(red[0] * (1.0f / 128.0f) + 1.1920928955078125e-07f);
    val *= rs;
    sv[d] = val;
    __syncthreads();
    float out;
    if (d < 64) {
      int i = (d < 32) ? d : (d - 32);
      // inv_freq = 10000^(-i/32) ; theta = s * inv_freq
      float theta = (float)s * __expf(-(float)i * (9.210340371976184f / 32.0f));
      float c = __cosf(theta), sn = __sinf(theta);
      if (d < 32) out =  sv[d] * c + sv[d + 32] * sn;
      else        out = -sv[d - 32] * sn + sv[d] * c;
    } else {
      out = val;
    }
    if (slot < 16) {
      qh[(((size_t)b * 16 + slot) * 1024 + s) * 128 + d] = (bf16_t)out;
    } else {
      int kv = slot - 16;
      kh[(((size_t)b * 4 + kv) * 1024 + s) * 128 + d] = (bf16_t)out;
    }
  } else {
    int kv = slot - 20;
    vf[(((size_t)b * 4 + kv) * 1024 + s) * 128 + d] = val;
    vT[(((size_t)b * 4 + kv) * 128 + d) * 1024 + s] = (bf16_t)val;
  }
}

// ---------------------------------------------------------------------------
// Flash attention. Block = 4 waves; each wave owns 16 query rows
// (block covers 64). Per step: 32 keys staged into LDS with
// global_load_async_to_lds_b128 (ASYNCcnt), QK = 2 score tiles x 4 WMMAs
// from conflict-free padded LDS, online softmax (16-lane shfl reductions per
// the C/D layout), P staged through LDS into A-fragment layout, PV = 8 WMMAs.
// Causal handled with -10000 bias (matches reference) + causal loop bound.
// ---------------------------------------------------------------------------
__global__ void __launch_bounds__(128)
attn_kernel(const bf16_t* __restrict__ qh, const bf16_t* __restrict__ kh,
            const bf16_t* __restrict__ vT, float* __restrict__ y) {
  const int tid  = threadIdx.x;
  const int lane = tid & 31;
  const int wave = tid >> 5;
  const int l15  = lane & 15;
  const int hi   = lane >> 4;
  const int S = 1024;
  const int q0 = blockIdx.x * 64;
  const int h  = blockIdx.y;
  const int b  = blockIdx.z;
  const int qw = q0 + wave * 16;

  const size_t qbase = ((size_t)b * 16 + h) * S * 128;          // [B,16,S,128]
  const size_t kbase = ((size_t)b * 4 + (h >> 2)) * S * 128;    // [B,4,S,128]
  const size_t vbase = ((size_t)b * 4 + (h >> 2)) * 128 * (size_t)S; // [B,4,128,S]

  // K tile: 32 keys x 128 dims, padded to 136 bf16/row (68 DWORDs -> banks
  // rotate by 4 per row; 16-lane fragment reads are conflict-free).
  __shared__ __attribute__((aligned(16))) bf16_t kt[32][136];
  __shared__ __attribute__((aligned(16))) bf16_t pbuf[4][16][40];

  v16bf qf[4];
#pragma unroll
  for (int dk = 0; dk < 4; ++dk) {
    const bf16_t* qp = qh + qbase + (size_t)(qw + l15) * 128 + dk * 32 + hi * 8;
    qf[dk] = cat8(*(const v8bf*)qp, *(const v8bf*)(qp + 16));
  }

  v8f oacc[8];
#pragma unroll
  for (int t = 0; t < 8; ++t) oacc[t] = vzero8();
  float mrow[8], lrow[8];
#pragma unroll
  for (int i = 0; i < 8; ++i) { mrow[i] = -1e30f; lrow[i] = 0.0f; }

  const float scale = 0.08838834764831845f;  // 1/sqrt(128)
  const int jend = q0 + 64;                  // causal bound (uniform per block)

  for (int j = 0; j < jend; j += 32) {
    // ---- stage K tile into LDS via async copies ----
    __syncthreads();  // protect kt against readers of previous step
#pragma unroll
    for (int q = 0; q < 4; ++q) {
      int c   = tid + 128 * q;   // 512 x 16B chunks
      int row = c >> 4;
      int cc  = (c & 15) * 8;
      unsigned lo = (unsigned)(uintptr_t)(&kt[row][cc]);
      async_g2l_b128(lo, kh + kbase + (size_t)(j + row) * 128 + cc);
    }
    wait_asynccnt0();
    __syncthreads();  // kt visible to all waves

    // ---- scores: two 16x16 tiles from LDS K ----
    v8f st[2];
#pragma unroll
    for (int t = 0; t < 2; ++t) {
      st[t] = vzero8();
#pragma unroll
      for (int dk = 0; dk < 4; ++dk) {
        const bf16_t* kp = &kt[16 * t + l15][dk * 32 + hi * 8];
        v16bf bk = cat8(*(const v8bf*)kp, *(const v8bf*)(kp + 16));
        st[t] = __builtin_amdgcn_wmma_f32_16x16x32_bf16(
            false, qf[dk], false, bk, (short)0, st[t], false, false);
      }
    }
    // ---- scale + causal bias ----
#pragma unroll
    for (int t = 0; t < 2; ++t) {
      int col = j + 16 * t + l15;
#pragma unroll
      for (int i = 0; i < 8; ++i) {
        int row = qw + 8 * hi + i;
        st[t][i] = st[t][i] * scale + ((col > row) ? -10000.0f : 0.0f);
      }
    }
    // ---- online softmax (row stats live in a 16-lane group) ----
    float mnew[8], alpha[8], psum[8];
#pragma unroll
    for (int i = 0; i < 8; ++i) {
      float v = fmaxf(st[0][i], st[1][i]);
#pragma unroll
      for (int mk = 1; mk < 16; mk <<= 1) v = fmaxf(v, __shfl_xor(v, mk, 32));
      mnew[i]  = fmaxf(mrow[i], v);
      alpha[i] = __expf(mrow[i] - mnew[i]);
    }
#pragma unroll
    for (int i = 0; i < 8; ++i) {
      float p0 = __expf(st[0][i] - mnew[i]);
      float p1 = __expf(st[1][i] - mnew[i]);
      st[0][i] = p0;
      st[1][i] = p1;
      float v = p0 + p1;
#pragma unroll
      for (int mk = 1; mk < 16; mk <<= 1) v += __shfl_xor(v, mk, 32);
      psum[i] = v;
    }
#pragma unroll
    for (int i = 0; i < 8; ++i) {
      lrow[i] = lrow[i] * alpha[i] + psum[i];
      mrow[i] = mnew[i];
    }
#pragma unroll
    for (int t = 0; t < 8; ++t)
#pragma unroll
      for (int i = 0; i < 8; ++i) oacc[t][i] *= alpha[i];

    // ---- P: C/D layout -> A-fragment layout via LDS ----
    __syncthreads();
#pragma unroll
    for (int t = 0; t < 2; ++t)
#pragma unroll
      for (int i = 0; i < 8; ++i)
        pbuf[wave][8 * hi + i][16 * t + l15] = (bf16_t)st[t][i];
    __syncthreads();
    const bf16_t* pp = &pbuf[wave][l15][hi * 8];
    v16bf pf = cat8(*(const v8bf*)pp, *(const v8bf*)(pp + 16));

    // ---- PV: 8 output d-tiles (V^T contiguous from global/L2) ----
#pragma unroll
    for (int dt = 0; dt < 8; ++dt) {
      const bf16_t* vp = vT + vbase + (size_t)(dt * 16 + l15) * S + j + hi * 8;
      v16bf bv = cat8(*(const v8bf*)vp, *(const v8bf*)(vp + 16));
      oacc[dt] = __builtin_amdgcn_wmma_f32_16x16x32_bf16(
          false, pf, false, bv, (short)0, oacc[dt], false, false);
    }
  }

  // ---- normalize and store y[B,16,S,128] fp32 ----
#pragma unroll
  for (int dt = 0; dt < 8; ++dt)
#pragma unroll
    for (int i = 0; i < 8; ++i) {
      int row = qw + 8 * hi + i;
      int col = dt * 16 + l15;
      y[qbase + (size_t)row * 128 + col] = oacc[dt][i] / lrow[i];
    }
}

// ---------------------------------------------------------------------------
// y -= (y.v / max(v.v,1e-6)) * v   per (b,h,s) row, then transpose to
// [B,S,H*HD] and cast to bf16 for the output projection.
// ---------------------------------------------------------------------------
__global__ void __launch_bounds__(128)
proj_remove_kernel(const float* __restrict__ y, const float* __restrict__ vf,
                   bf16_t* __restrict__ yb) {
  const int s = blockIdx.x, h = blockIdx.y, b = blockIdx.z;
  const int d = threadIdx.x;
  __shared__ float r1[128];
  __shared__ float r2[128];
  size_t yi = (((size_t)b * 16 + h) * 1024 + s) * 128 + d;
  size_t vi = (((size_t)b * 4 + (h >> 2)) * 1024 + s) * 128 + d;
  float yv = y[yi], vv = vf[vi];
  r1[d] = yv * vv;
  r2[d] = vv * vv;
  __syncthreads();
  for (int st = 64; st > 0; st >>= 1) {
    if (d < st) { r1[d] += r1[d + st]; r2[d] += r2[d + st]; }
    __syncthreads();
  }
  float proj = r1[0] / fmaxf(r2[0], 1e-6f);
  float out = yv - proj * vv;
  yb[((size_t)(b * 1024 + s)) * 2048 + (size_t)h * 128 + d] = (bf16_t)out;
}

// ---------------------------------------------------------------------------
// Host-side orchestration
// ---------------------------------------------------------------------------
extern "C" void kernel_launch(void* const* d_in, const int* in_sizes, int n_in,
                              void* d_out, int out_size, void* d_ws,
                              size_t ws_size, hipStream_t stream) {
  (void)in_sizes; (void)n_in; (void)out_size; (void)ws_size;
  const float* x     = (const float*)d_in[0];
  // d_in[1] = attn_mask (causal tril by construction) - not needed
  const float* Wq    = (const float*)d_in[2];
  const float* Wk    = (const float*)d_in[3];
  const float* Wv    = (const float*)d_in[4];
  const float* Wproj = (const float*)d_in[5];
  float* out = (float*)d_out;

  // workspace carve (all sizes in bytes, 256B-aligned by construction)
  char* p = (char*)d_ws;
  bf16_t* xb     = (bf16_t*)p; p += 16777216;  // x bf16   [4096,2048]; reused as yb
  bf16_t* wqkvb  = (bf16_t*)p; p += 12582912;  // [3072,2048] bf16 (Wq|Wk|Wv)
  bf16_t* wprojb = (bf16_t*)p; p += 8388608;   // [2048,2048] bf16
  float*  qkv    = (float*)p;  p += 50331648;  // [4096,3072] fp32; reused as y
  bf16_t* qh     = (bf16_t*)p; p += 16777216;  // [4,16,1024,128] bf16
  bf16_t* kh     = (bf16_t*)p; p += 4194304;   // [4,4,1024,128] bf16
  bf16_t* vT     = (bf16_t*)p; p += 4194304;   // [4,4,128,1024] bf16
  float*  vf     = (float*)p;  p += 8388608;   // [4,4,1024,128] fp32
  bf16_t* yb = xb;      // alias: x bf16 dead after QKV GEMM
  float*  y  = qkv;     // alias: qkv fp32 dead after norm/rope

  auto cgrid = [](long n) { return dim3((unsigned)((n / 4 + 255) / 256)); };

  // 1) casts
  cast_f32_bf16<<<cgrid(8388608), 256, 0, stream>>>(x, xb, 8388608L);
  cast_f32_bf16<<<cgrid(4194304), 256, 0, stream>>>(Wq, wqkvb, 4194304L);
  cast_f32_bf16<<<cgrid(1048576), 256, 0, stream>>>(Wk, wqkvb + 4194304, 1048576L);
  cast_f32_bf16<<<cgrid(1048576), 256, 0, stream>>>(Wv, wqkvb + 5242880, 1048576L);
  cast_f32_bf16<<<cgrid(4194304), 256, 0, stream>>>(Wproj, wprojb, 4194304L);

  // 2) fused QKV projection: [4096,3072] = xb[4096,2048] * wqkvb[3072,2048]^T
  gemm_bf16_nt<<<dim3(3072 / 128, 4096 / 128), 128, 0, stream>>>(
      xb, wqkvb, qkv, 4096, 3072, 2048);

  // 3) rms-norm + rope + layout shuffle
  normrope_kernel<<<dim3(4096, 24), 128, 0, stream>>>(qkv, qh, kh, vT, vf);

  // 4) flash attention (async-LDS K staging + WMMA)
  attn_kernel<<<dim3(16, 16, 4), 128, 0, stream>>>(qh, kh, vT, y);

  // 5) project-out-v + transpose + cast
  proj_remove_kernel<<<dim3(1024, 16, 4), 128, 0, stream>>>(y, vf, yb);

  // 6) output projection: out[4096,2048] = yb[4096,2048] * wprojb[2048,2048]^T
  gemm_bf16_nt<<<dim3(2048 / 128, 4096 / 128), 128, 0, stream>>>(
      yb, wprojb, out, 4096, 2048, 2048);
}
